// WindowAttention3D_57827439673427
// MI455X (gfx1250) — matrix operand order repaired and checked by reading, not compiled, not run on hardware
//
#include <hip/hip_runtime.h>
#include <hip/hip_bf16.h>
#include <stdint.h>

typedef __attribute__((ext_vector_type(16))) _Float16 v16h;
typedef __attribute__((ext_vector_type(8)))  _Float16 h8;
typedef __attribute__((ext_vector_type(8)))  float    v8f;
typedef __attribute__((ext_vector_type(2)))  _Float16 h2;
typedef __attribute__((ext_vector_type(4)))  uint32_t u32x4;
typedef __attribute__((ext_vector_type(8)))  uint32_t u32x8;

#define C_DIM   128
#define T_DIM   64
#define NHEADS  8
#define LN_EPS  1e-5f

// global strides for [B=2, C=128, 48,48,48] fp32
#define S_H 48
#define S_D (48*48)
#define S_C (48*48*48)
#define S_B (128*48*48*48)

// LDS strides (chosen for 16B-aligned b128 fragment loads + bank spread)
#define XW2_LD 68   // fp32 TDM staging [128 ch][64 tok + 4 pad dwords]
#define XN_LD  136  // f16 [64][136] (byte stride 272 = 17*16)
#define VT_LD  72   // f16 V^T [128 ch][64 tok + 8 pad] (byte stride 144 = 9*16)
#define P_LD   72   // f16 per-head attn probs [64][72]
#define OD_LD  64   // fp32 dense out staging [128 ch][64 tok] (TDM store source)

// LDS region layout (bytes); region R1 is a union of {xw fp32, P f16, outd fp32}
#define R1_BYTES   (NHEADS * T_DIM * P_LD * 2)          // 73728 (covers 128*68*4=34816, 128*64*4=32768)
#define XN_BYTES   (T_DIM * XN_LD * 2)                  // 17408
#define VT_BYTES   (C_DIM * VT_LD * 2)                  // 18432
#define OFF_XN     R1_BYTES
#define OFF_QKV    (OFF_XN + XN_BYTES)
#define SMEM_BYTES (OFF_QKV + 2 * XN_BYTES + VT_BYTES)  // 144384

// f16 weight cache in d_ws
#define WQ_ELEMS (384 * 128)
#define WO_ELEMS (128 * 128)

static __device__ __forceinline__ v8f wmma_f16(const v16h& a, const v16h& b, const v8f& c) {
  return __builtin_amdgcn_wmma_f32_16x16x32_f16(false, a, false, b, (short)0, c, false, false);
}
static __device__ __forceinline__ v16h cat16(h8 a, h8 b) {
  return __builtin_shufflevector(a, b, 0,1,2,3,4,5,6,7,8,9,10,11,12,13,14,15);
}
// A-operand fragment (row-major source): halves p[0..7] ++ p[16..23]  (2x b128)
static __device__ __forceinline__ v16h load_afrag(const _Float16* p) {
  return cat16(*(const h8*)p, *(const h8*)(p + 16));
}
// B-operand fragment from 16 contiguous halves (2x b128)
static __device__ __forceinline__ v16h load_bfrag16(const _Float16* p) {
  return cat16(*(const h8*)p, *(const h8*)(p + 8));
}

// ---- pre-pass: convert weights fp32 -> f16 row-major, once for all windows ----
__global__ void cvt_weights_kernel(const float* __restrict__ wq,
                                   const float* __restrict__ wo,
                                   _Float16* __restrict__ dst) {
  int i = blockIdx.x * 256 + threadIdx.x;          // 0 .. 65535
  float v = (i < WQ_ELEMS) ? wq[i] : wo[i - WQ_ELEMS];
  dst[i] = (_Float16)v;
}

__global__ __launch_bounds__(256)
void win_attn3d_kernel(const float* __restrict__ x,
                       const _Float16* __restrict__ wqh,   // [384][128] f16
                       const float* __restrict__ b_qkv,
                       const _Float16* __restrict__ woh,   // [128][128] f16
                       const float* __restrict__ b_out,
                       const float* __restrict__ gamma,
                       const float* __restrict__ beta,
                       float* __restrict__ out) {
  extern __shared__ char smem[];
  float*    xwc  = (float*)(smem);               // phase 1-2: [c][68] fp32 (TDM dest)
  _Float16* Pb   = (_Float16*)(smem);            // phase 4 (after xw dead)
  float*    outd = (float*)(smem);               // phase 5: dense [c][64] fp32 (TDM store src)
  _Float16* xn   = (_Float16*)(smem + OFF_XN);   // normalized tokens; reused as O buffer [t][c]
  _Float16* qld  = (_Float16*)(smem + OFF_QKV);  // [t][c]
  _Float16* kld  = qld + T_DIM * XN_LD;          // [t][c]
  _Float16* vtd  = kld + T_DIM * XN_LD;          // V^T [c][t], stride VT_LD

  const int tid  = threadIdx.x;
  const int lane = tid & 31;
  const int wv   = tid >> 5;     // wave id 0..7
  const int nL   = lane & 15;
  const int hi   = lane >> 4;

  // ---- window decode: blockIdx.x in [0, 2*12*12*12) ----
  int win = blockIdx.x;
  int bI  = win / 1728; int rem = win - bI * 1728;
  int wd  = rem / 144;  rem -= wd * 144;
  int wh  = rem / 12;   int wwi = rem - wh * 12;
  const long base = (long)bI * S_B + (long)(wd * 4) * S_D + (wh * 4) * S_H + wwi * 4;

  // L2 hints for the shared f16 weight cache
  if (tid < 64) {
    __builtin_prefetch(wqh + tid * 768, 0, 1);
    __builtin_prefetch(woh + tid * 256, 0, 1);
  }

  // ========== Phase 1: TDM DMA of the 4x4x4x128 window into LDS ==========
  // D#: 4D tile (x=W:4, y=H:4, z=D:4, dim3=C:128), data_size=4B,
  // LDS pad 4 dwords every 64 dwords -> LDS layout [c][64+4] fp32.
  if (wv == 0) {
    uint32_t lds_base = (uint32_t)(uintptr_t)(void*)xwc;
    unsigned long long ga = (unsigned long long)(uintptr_t)(x + base);
    u32x4 g0;
    g0[0] = 1u;                                          // count=1, user descriptor
    g0[1] = lds_base;                                    // lds_addr
    g0[2] = (uint32_t)ga;                                // global_addr[31:0]
    g0[3] = (uint32_t)((ga >> 32) & 0x1FFFFFFu) | (2u << 30);  // global_addr[56:32] | type=2
    u32x8 g1;
    g1[0] = (2u << 16) | (1u << 20) | (5u << 22) | (3u << 25); // 4B, pad_en, every 64 dw, +4 dw
    g1[1] = (48u << 16);                                 // tensor_dim0 = 48
    g1[2] = (48u << 16);                                 // tensor_dim1 = 48
    g1[3] = (4u << 16);                                  // tile_dim0 = 4
    g1[4] = 4u | (4u << 16);                             // tile_dim1 = 4, tile_dim2 = 4
    g1[5] = 48u;                                         // tensor_dim0_stride = 48
    g1[6] = (2304u & 0xffffu) << 16;                     // tensor_dim1_stride lo = 2304
    g1[7] = 0u;
    u32x4 g2;
    g2[0] = 48u; g2[1] = 128u; g2[2] = 110592u; g2[3] = (128u << 16); // dims 2/3, dim2_stride, tile_dim3
    u32x4 g3;
    g3[0] = 110592u; g3[1] = 0u; g3[2] = 0u; g3[3] = 0u; // tensor_dim3_stride
    asm volatile("tensor_load_to_lds %0, %1, %2, %3"
                 :: "s"(g0), "s"(g1), "s"(g2), "s"(g3) : "memory");
    __builtin_amdgcn_s_wait_tensorcnt(0);
  }
  __syncthreads();

  // ========== Phase 2: LayerNorm over C (4 lanes per token), reading [c][68] ==========
  {
    int t = tid >> 2, qp = tid & 3, c0 = qp * 32;
    float s = 0.f, s2 = 0.f;
    #pragma unroll
    for (int c = 0; c < 32; ++c) { float v = xwc[(c0 + c) * XW2_LD + t]; s += v; s2 += v * v; }
    s  += __shfl_xor(s, 1, 32);  s  += __shfl_xor(s, 2, 32);
    s2 += __shfl_xor(s2, 1, 32); s2 += __shfl_xor(s2, 2, 32);
    float mu   = s * (1.0f / 128.0f);
    float var  = s2 * (1.0f / 128.0f) - mu * mu;
    float rstd = rsqrtf(var + LN_EPS);
    #pragma unroll
    for (int c = 0; c < 32; ++c) {
      int cc = c0 + c;
      float v = (xwc[cc * XW2_LD + t] - mu) * rstd * gamma[cc] + beta[cc];
      xn[t * XN_LD + cc] = (_Float16)v;
    }
  }
  __syncthreads();

  // ========== Phase 3: fused QKV GEMM [64x128] @ [128x384], f16 weights ==========
  for (int g = 0; g < 3; ++g) {
    int nt = wv * 3 + g;            // 0..23
    int e  = nt * 16 + nL;          // output feature 0..383
    float bias = b_qkv[e];
    v8f acc[4] = {};
    for (int ks = 0; ks < 4; ++ks) {
      // B fragment = 16 contiguous halves W[e][ks*32+hi*16 .. +15] : one 32B load
      v16h bf = *(const v16h*)(wqh + e * C_DIM + ks * 32 + hi * 16);
      #pragma unroll
      for (int mt = 0; mt < 4; ++mt) {
        v16h af = load_bfrag16(xn + (mt * 16 + nL) * XN_LD + ks * 32 + hi * 16);
        acc[mt] = wmma_f16(af, bf, acc[mt]);
      }
    }
    if (nt < 16) {                  // q (scaled by 1/sqrt(dh)) and k, layout [t][c]
      _Float16* dst = (nt < 8) ? qld : kld;
      int c = (nt & 7) * 16 + nL;
      float scale = (nt < 8) ? 0.25f : 1.0f;
      #pragma unroll
      for (int mt = 0; mt < 4; ++mt)
        #pragma unroll
        for (int v = 0; v < 8; ++v)
          dst[(mt * 16 + hi * 8 + v) * XN_LD + c] = (_Float16)((acc[mt][v] + bias) * scale);
    } else {                        // v transposed: [c][t], contiguous along tokens
      int c = (nt - 16) * 16 + nL;
      #pragma unroll
      for (int mt = 0; mt < 4; ++mt)
        #pragma unroll
        for (int v = 0; v < 8; v += 2) {
          h2 pv;
          pv.x = (_Float16)(acc[mt][v]     + bias);
          pv.y = (_Float16)(acc[mt][v + 1] + bias);
          *(h2*)(vtd + c * VT_LD + mt * 16 + hi * 8 + v) = pv;
        }
    }
  }
  __syncthreads();

  // ========== Phase 4: attention, one head per wave ==========
  {
    const int h = wv;
    _Float16* Ph = Pb + h * T_DIM * P_LD;
    const h8 z8 = {};

    // B fragments for S = q.k^T : B[kk][n] = k[token n][h*16+kk], kk<16 valid, rest zero
    v16h bk[4];
    #pragma unroll
    for (int nt = 0; nt < 4; ++nt) {
      const _Float16* kp = kld + (nt * 16 + nL) * XN_LD + h * 16;
      h8 b0 = *(const h8*)kp;
      h8 b1 = *(const h8*)(kp + 8);
      bk[nt] = cat16(hi ? b1 : b0, z8);
    }
    #pragma unroll
    for (int mt = 0; mt < 4; ++mt) {
      // A[m][kk] = q[m][h*16+kk], kk<16 valid -> low 8 regs only
      h8 a0 = *(const h8*)(qld + (mt * 16 + nL) * XN_LD + h * 16 + hi * 8);
      v16h aq = cat16(a0, z8);
      v8f s0 = {}, s1 = {}, s2 = {}, s3 = {};
      s0 = wmma_f16(aq, bk[0], s0);
      s1 = wmma_f16(aq, bk[1], s1);
      s2 = wmma_f16(aq, bk[2], s2);
      s3 = wmma_f16(aq, bk[3], s3);
      // row-wise softmax: a row lives in one 16-lane half
      #pragma unroll
      for (int v = 0; v < 8; ++v) {
        float a = s0[v], b2 = s1[v], c2 = s2[v], d2 = s3[v];
        float m = fmaxf(fmaxf(a, b2), fmaxf(c2, d2));
        #pragma unroll
        for (int off = 8; off; off >>= 1) m = fmaxf(m, __shfl_xor(m, off, 32));
        a = __expf(a - m); b2 = __expf(b2 - m); c2 = __expf(c2 - m); d2 = __expf(d2 - m);
        float sum = a + b2 + c2 + d2;
        #pragma unroll
        for (int off = 8; off; off >>= 1) sum += __shfl_xor(sum, off, 32);
        float inv = 1.0f / sum;
        int row = mt * 16 + hi * 8 + v;
        Ph[row * P_LD +  0 + nL] = (_Float16)(a  * inv);
        Ph[row * P_LD + 16 + nL] = (_Float16)(b2 * inv);
        Ph[row * P_LD + 32 + nL] = (_Float16)(c2 * inv);
        Ph[row * P_LD + 48 + nL] = (_Float16)(d2 * inv);
      }
    }
    // O = P @ V : M=64, N=16, K=64 (2 k-steps); V^T gives contiguous B fragments
    #pragma unroll
    for (int mt = 0; mt < 4; ++mt) {
      v8f acc = {};
      #pragma unroll
      for (int ks = 0; ks < 2; ++ks) {
        v16h af = load_bfrag16(Ph + (mt * 16 + nL) * P_LD + ks * 32 + hi * 16);
        v16h bf = load_bfrag16(vtd + (h * 16 + nL) * VT_LD + ks * 32 + hi * 16);
        acc = wmma_f16(af, bf, acc);
      }
      #pragma unroll
      for (int v = 0; v < 8; ++v)   // O into xn region [t][c] (normalized input is dead)
        xn[(mt * 16 + hi * 8 + v) * XN_LD + h * 16 + nL] = (_Float16)acc[v];
    }
  }
  __syncthreads();

  // ========== Phase 5: transposed out projection OUT^T = W_out @ O^T ==========
  // C-tile rows = channels, cols = tokens -> dense [c][64] LDS store is conflict-free
  {
    int erow = wv * 16;             // each wave owns 16 output channels
    float bo[8];
    #pragma unroll
    for (int v = 0; v < 8; ++v) bo[v] = b_out[erow + hi * 8 + v];
    v8f acc[4] = {};
    for (int ks = 0; ks < 4; ++ks) {
      // A[m=e][k=c] from f16 weight rows (2x global b128)
      v16h af = load_bfrag16(woh + (erow + nL) * C_DIM + ks * 32 + hi * 16);
      #pragma unroll
      for (int mt = 0; mt < 4; ++mt) {
        // B[k=c][n=t] = O[t][c]: 16 contiguous halves per lane
        v16h bf = load_bfrag16(xn + (mt * 16 + nL) * XN_LD + ks * 32 + hi * 16);
        acc[mt] = wmma_f16(af, bf, acc[mt]);
      }
    }
    #pragma unroll
    for (int mt = 0; mt < 4; ++mt)
      #pragma unroll
      for (int v = 0; v < 8; ++v)
        outd[(erow + hi * 8 + v) * OD_LD + mt * 16 + nL] = acc[mt][v] + bo[v];
  }
  __syncthreads();

  // ========== Phase 6: TDM DMA store of the dense [c][64] tile back to global ==========
  if (wv == 0) {
    uint32_t lds_base = (uint32_t)(uintptr_t)(void*)outd;
    unsigned long long ga = (unsigned long long)(uintptr_t)(out + base);
    u32x4 g0;
    g0[0] = 1u;
    g0[1] = lds_base;
    g0[2] = (uint32_t)ga;
    g0[3] = (uint32_t)((ga >> 32) & 0x1FFFFFFu) | (2u << 30);
    u32x8 g1;
    g1[0] = (2u << 16);                                  // data_size=4B, no pad (stores don't de-pad)
    g1[1] = (48u << 16);
    g1[2] = (48u << 16);
    g1[3] = (4u << 16);
    g1[4] = 4u | (4u << 16);
    g1[5] = 48u;
    g1[6] = (2304u & 0xffffu) << 16;
    g1[7] = 0u;
    u32x4 g2;
    g2[0] = 48u; g2[1] = 128u; g2[2] = 110592u; g2[3] = (128u << 16);
    u32x4 g3;
    g3[0] = 110592u; g3[1] = 0u; g3[2] = 0u; g3[3] = 0u;
    asm volatile("tensor_store_from_lds %0, %1, %2, %3"
                 :: "s"(g0), "s"(g1), "s"(g2), "s"(g3) : "memory");
    __builtin_amdgcn_s_wait_tensorcnt(0);
  }
}

extern "C" void kernel_launch(void* const* d_in, const int* in_sizes, int n_in,
                              void* d_out, int out_size, void* d_ws, size_t ws_size,
                              hipStream_t stream) {
  const float* x     = (const float*)d_in[0];
  const float* w_qkv = (const float*)d_in[1];
  const float* b_qkv = (const float*)d_in[2];
  const float* w_out = (const float*)d_in[3];
  const float* b_out = (const float*)d_in[4];
  const float* gamma = (const float*)d_in[5];
  const float* beta  = (const float*)d_in[6];
  float* out = (float*)d_out;

  _Float16* wqh = (_Float16*)d_ws;                 // 384*128 f16
  _Float16* woh = wqh + WQ_ELEMS;                  // 128*128 f16

  (void)hipFuncSetAttribute((const void*)win_attn3d_kernel,
                            hipFuncAttributeMaxDynamicSharedMemorySize, SMEM_BYTES);

  // one-shot weight conversion (L2-resident thereafter)
  hipLaunchKernelGGL(cvt_weights_kernel, dim3((WQ_ELEMS + WO_ELEMS) / 256), dim3(256), 0, stream,
                     w_qkv, w_out, wqh);

  dim3 grid(2 * 12 * 12 * 12);   // one workgroup per 4x4x4 window
  dim3 block(256);               // 8 wave32
  hipLaunchKernelGGL(win_attn3d_kernel, grid, block, SMEM_BYTES, stream,
                     x, wqh, b_qkv, woh, b_out, gamma, beta, out);
}
